// MultiHeadAttention_1889785610416
// MI455X (gfx1250) — matrix operand order repaired and checked by reading, compiled, not run on hardware
//
#include <hip/hip_runtime.h>
#include <hip/hip_bf16.h>
#include <stdint.h>

// Problem constants (must match reference)
#define B_   4
#define L_   2048
#define DX_  1024
#define H_   16
#define DKV_ 64
#define EPS_ 1e-5f

typedef __attribute__((ext_vector_type(16))) _Float16 v16h;
typedef __attribute__((ext_vector_type(8)))  float    v8f;
typedef __attribute__((ext_vector_type(4)))  unsigned int u32x4;
typedef __attribute__((ext_vector_type(8)))  int      i32x8;
typedef __attribute__((ext_vector_type(4)))  int      i32x4;

static __device__ __forceinline__ v8f wmma_f16(v16h a, v16h b, v8f c) {
  // D = A(16x32 f16) * B(32x16 f16) + C(16x16 f32)
  return __builtin_amdgcn_wmma_f32_16x16x32_f16(false, a, false, b, (short)0, c,
                                                false, false);
}

// ---------------- fragment loaders (wave32, contiguous k = 16*half + e per lane) ----------------
static __device__ __forceinline__ v16h frag_a_f32(const float* __restrict__ A, int lda) {
  const int lane = threadIdx.x & 31, half = lane >> 4, row = lane & 15;
  const float* p = A + (size_t)row * lda;
  v16h a;
#pragma unroll
  for (int e = 0; e < 16; ++e) {
    const int kk = (half << 4) + e;
    a[e] = (_Float16)p[kk];
  }
  return a;
}

static __device__ __forceinline__ v16h frag_a_f16(const _Float16* __restrict__ A, int lda) {
  const int lane = threadIdx.x & 31, half = lane >> 4, row = lane & 15;
  const _Float16* p = A + (size_t)row * lda;
  v16h a;
#pragma unroll
  for (int e = 0; e < 16; ++e) {
    const int kk = (half << 4) + e;
    a[e] = p[kk];
  }
  return a;
}

static __device__ __forceinline__ v16h frag_b_f16(const _Float16* __restrict__ Bm, int ldb) {
  const int lane = threadIdx.x & 31, half = lane >> 4, col = lane & 15;
  v16h b;
#pragma unroll
  for (int e = 0; e < 16; ++e) {
    const int kk = (half << 4) + e;
    b[e] = Bm[(size_t)kk * ldb + col];
  }
  return b;
}

// B[k][n] = S[n*lds + k]  (transposed-layout source, 16 contiguous f16 per lane)
static __device__ __forceinline__ v16h frag_bT_f16(const _Float16* __restrict__ S, int lds) {
  const int lane = threadIdx.x & 31, half = lane >> 4, col = lane & 15;
  const _Float16* p = S + (size_t)col * lds + (half << 4);
  v16h b;
#pragma unroll
  for (int e = 0; e < 16; ++e) b[e] = p[e];
  return b;
}

// ---------------- Tensor Data Mover: 2D f16 tile -> LDS ----------------
// Loads a tile_dim1=64 rows x tile_dim0=32 f16 tile from a row-major
// [1024 x 1024] f16 tensor (row stride 1024 elems) into LDS (contiguous).
static __device__ __forceinline__ void tdm_load_w_tile(const _Float16* gptr,
                                                       unsigned lds_byte_addr) {
  const unsigned long long ga = (unsigned long long)(uintptr_t)gptr;
  // D# group0: count=1 | lds_addr | global_addr[56:0] | type=2 (bits 127:126)
  u32x4 g0 = {1u, lds_byte_addr, (unsigned)ga,
              (unsigned)(ga >> 32) | (2u << 30)};
  // D# group1: data_size=1(2B) @bits17:16 ; tensor_dim0=1024 @79:48 ;
  // tensor_dim1=1024 @111:80 ; tile_dim0=32 @127:112 ; tile_dim1=64 @143:128 ;
  // tensor_dim0_stride=1024 @207:160
  i32x8 g1 = {(int)(1u << 16),    // dword0: data_size=2B
              (int)(1024u << 16), // dword1: tensor_dim0 low16 -> bits 63:48
              (int)(1024u << 16), // dword2: tensor_dim1 low16 -> bits 95:80... 111:80
              (int)(32u << 16),   // dword3: tile_dim0 -> bits 127:112
              64,                 // dword4: tile_dim1 -> bits 143:128
              1024,               // dword5: tensor_dim0_stride[31:0]
              0, 0};
  i32x4 gz = {0, 0, 0, 0};
#if defined(__clang_major__) && (__clang_major__ >= 23)
  i32x8 gz8 = {0, 0, 0, 0, 0, 0, 0, 0};
  __builtin_amdgcn_tensor_load_to_lds(g0, g1, gz, gz, gz8, 0);
#else
  __builtin_amdgcn_tensor_load_to_lds(g0, g1, gz, gz, 0);
#endif
}

// ---------------- kernels ----------------

// WT[n][k] = (f16) W[k][n] for a 1024x1024 matrix; LDS-tiled transpose.
__global__ void transpose_w_f16(const float* __restrict__ W,
                                _Float16* __restrict__ WT) {
  __shared__ float tile[32][33];
  const int k0 = blockIdx.x * 32, n0 = blockIdx.y * 32;
  const int tx = threadIdx.x & 31, ty = threadIdx.x >> 5; // 32x8
#pragma unroll
  for (int j = 0; j < 32; j += 8)
    tile[ty + j][tx] = W[(size_t)(k0 + ty + j) * DX_ + n0 + tx];
  __syncthreads();
#pragma unroll
  for (int j = 0; j < 32; j += 8)
    WT[(size_t)(n0 + ty + j) * DX_ + k0 + tx] = (_Float16)tile[tx][ty + j];
}

// QKV projection: [8192 x 1024] = X @ W + bias, store f16.
// Workgroup = 8 waves sharing one 64-col weight slab: m0 = mblk*128 + wave*16.
// Per K-step the 64x32 f16 weight tile is DMA'd into LDS by the Tensor Data
// Mover (double-buffered, s_wait_tensorcnt), then each wave runs 4 WMMAs.
__global__ void __launch_bounds__(256) qkv_proj(
    const float* __restrict__ Q, const float* __restrict__ K,
    const float* __restrict__ V,
    const _Float16* __restrict__ WqT, const _Float16* __restrict__ WkT,
    const _Float16* __restrict__ WvT,
    const float* __restrict__ bq, const float* __restrict__ bk,
    const float* __restrict__ bv,
    _Float16* __restrict__ Qs, _Float16* __restrict__ Ks,
    _Float16* __restrict__ Vs) {
  const float* X; const _Float16* WT; const float* bias; _Float16* O;
  if (blockIdx.y == 0)      { X = Q; WT = WqT; bias = bq; O = Qs; }
  else if (blockIdx.y == 1) { X = K; WT = WkT; bias = bk; O = Ks; }
  else                      { X = V; WT = WvT; bias = bv; O = Vs; }

  __shared__ _Float16 wtile[2][64 * 32];       // [buf][n*32 + k], 4 KB each

  const int wave = threadIdx.x >> 5;
  const int mblk = blockIdx.x >> 4;            // 64 row-blocks of 128
  const int ng   = blockIdx.x & 15;            // 16 col-slabs of 64
  const int m0   = mblk * 128 + wave * 16;
  const int n0   = ng * 64;
  const int lane = threadIdx.x & 31, half = lane >> 4, col = lane & 15;

  const _Float16* Wslab = WT + (size_t)n0 * DX_;   // 64 rows of WT

  if (wave == 0)
    tdm_load_w_tile(Wslab, (unsigned)(uintptr_t)&wtile[0][0]);

  v8f acc0 = {}, acc1 = {}, acc2 = {}, acc3 = {};
  for (int kt = 0; kt < 32; ++kt) {
    const int k0 = kt << 5;
    if (wave == 0) {
      if (kt + 1 < 32) {
        tdm_load_w_tile(Wslab + ((kt + 1) << 5),
                        (unsigned)(uintptr_t)&wtile[(kt + 1) & 1][0]);
        __builtin_amdgcn_s_wait_tensorcnt((short)1);
      } else {
        __builtin_amdgcn_s_wait_tensorcnt((short)0);
      }
    }
    __syncthreads();                           // tile[kt&1] ready for all waves

    const float* Arow = X + (size_t)m0 * DX_ + k0;
    __builtin_prefetch(Arow + 32, 0, 1);       // next K tile of A
    v16h a = frag_a_f32(Arow, DX_);

    const int bi = kt & 1;
    v16h b0, b1, b2, b3;
#pragma unroll
    for (int e = 0; e < 16; ++e) {
      const int kk = (half << 4) + e;          // contiguous k within lane row
      b0[e] = wtile[bi][(col)      * 32 + kk];
      b1[e] = wtile[bi][(16 + col) * 32 + kk];
      b2[e] = wtile[bi][(32 + col) * 32 + kk];
      b3[e] = wtile[bi][(48 + col) * 32 + kk];
    }
    acc0 = wmma_f16(a, b0, acc0);
    acc1 = wmma_f16(a, b1, acc1);
    acc2 = wmma_f16(a, b2, acc2);
    acc3 = wmma_f16(a, b3, acc3);
    __syncthreads();                           // done reading before reuse
  }

  const v8f* accs[4] = {&acc0, &acc1, &acc2, &acc3};
#pragma unroll
  for (int t = 0; t < 4; ++t) {
    const float bb = bias[n0 + t * 16 + col];
#pragma unroll
    for (int r = 0; r < 8; ++r) {
      const int m = (half << 3) + r;
      O[(size_t)(m0 + m) * DX_ + n0 + t * 16 + col] =
          (_Float16)((*accs[t])[r] + bb);
    }
  }
}

// Fused scores + mask + softmax: one workgroup = one (b, h, 16-row) block.
// Wave w owns columns [w*256, w*256+256) in 16 f32 accumulators (128 VGPRs);
// row max/sum reduced across waves via LDS; att written exactly once.
__global__ void __launch_bounds__(256) fused_scores_softmax(
    const _Float16* __restrict__ Qs, const _Float16* __restrict__ Ks,
    const unsigned char* __restrict__ mask, float* __restrict__ att) {
  __shared__ float red[8][32][8];
  __shared__ float rowmax_s[16];
  __shared__ float rowsum_s[16];

  const int pair = blockIdx.y;                 // b*H + h
  const int b = pair >> 4, h = pair & 15;
  const int m0 = blockIdx.x << 4;
  const int wave = threadIdx.x >> 5;
  const int lane = threadIdx.x & 31, half = lane >> 4, col = lane & 15;
  const int c0 = wave << 8;

  const _Float16* Qb = Qs + ((size_t)b * L_ + m0) * DX_ + h * DKV_;
  const _Float16* Kb = Ks + (size_t)b * L_ * DX_ + h * DKV_;

  const v16h a0 = frag_a_f16(Qb + 0, DX_);
  const v16h a1 = frag_a_f16(Qb + 32, DX_);

  v8f acc[16];
#pragma unroll
  for (int t = 0; t < 16; ++t) {
    const int n0 = c0 + (t << 4);
    const _Float16* Kn = Kb + (size_t)n0 * DX_;
    v16h b0 = frag_bT_f16(Kn + 0, DX_);
    v16h b1 = frag_bT_f16(Kn + 32, DX_);
    v8f c = {};
    c = wmma_f16(a0, b0, c);
    c = wmma_f16(a1, b1, c);
    acc[t] = c;
  }

  const unsigned char* Mb = mask + ((size_t)b * L_ + m0) * L_;
  float mymax[8];
#pragma unroll
  for (int r = 0; r < 8; ++r) mymax[r] = -__builtin_inff();
#pragma unroll
  for (int t = 0; t < 16; ++t) {
    const int n = c0 + (t << 4) + col;
#pragma unroll
    for (int r = 0; r < 8; ++r) {
      const int m = (half << 3) + r;
      float s = acc[t][r] * 0.125f;            // 1/sqrt(64)
      if (Mb[(size_t)m * L_ + n]) s = -__builtin_inff();
      acc[t][r] = s;
      mymax[r] = fmaxf(mymax[r], s);
    }
  }
#pragma unroll
  for (int r = 0; r < 8; ++r) red[wave][lane][r] = mymax[r];
  __syncthreads();

  if (threadIdx.x < 16) {
    const int m = threadIdx.x, hh = m >> 3, rr = m & 7;
    float mx = -__builtin_inff();
    for (int w = 0; w < 8; ++w)
      for (int l = 0; l < 16; ++l)
        mx = fmaxf(mx, red[w][(hh << 4) + l][rr]);
    rowmax_s[m] = mx;
  }
  __syncthreads();

  float rmax[8];
#pragma unroll
  for (int r = 0; r < 8; ++r) rmax[r] = rowmax_s[(half << 3) + r];

  float mysum[8];
#pragma unroll
  for (int r = 0; r < 8; ++r) mysum[r] = 0.f;
#pragma unroll
  for (int t = 0; t < 16; ++t) {
#pragma unroll
    for (int r = 0; r < 8; ++r) {
      const float e = __expf(acc[t][r] - rmax[r]);
      acc[t][r] = e;
      mysum[r] += e;
    }
  }
#pragma unroll
  for (int r = 0; r < 8; ++r) red[wave][lane][r] = mysum[r];
  __syncthreads();

  if (threadIdx.x < 16) {
    const int m = threadIdx.x, hh = m >> 3, rr = m & 7;
    float sm = 0.f;
    for (int w = 0; w < 8; ++w)
      for (int l = 0; l < 16; ++l)
        sm += red[w][(hh << 4) + l][rr];
    rowsum_s[m] = sm;
  }
  __syncthreads();

  float rinv[8];
#pragma unroll
  for (int r = 0; r < 8; ++r) rinv[r] = 1.f / rowsum_s[(half << 3) + r];

  float* Ab = att + (((size_t)(h * B_ + b)) * L_ + m0) * L_ + c0;
#pragma unroll
  for (int t = 0; t < 16; ++t) {
#pragma unroll
    for (int r = 0; r < 8; ++r) {
      const int m = (half << 3) + r;
      Ab[(size_t)m * L_ + (t << 4) + col] = acc[t][r] * rinv[r];
    }
  }
}

// ctx[b,m,h,:] = sum_k att[h*B+b,m,k] * Vs[b,k,h,:]
__global__ void attv_kernel(const float* __restrict__ att,
                            const _Float16* __restrict__ Vs,
                            _Float16* __restrict__ ctx) {
  const int pair = blockIdx.y;
  const int b = pair >> 4, h = pair & 15;
  const int wave = threadIdx.x >> 5;
  const int mt = blockIdx.x * 8 + wave;        // 0..127
  const int m0 = mt << 4;

  const float*    Ab = att + (((size_t)(h * B_ + b)) * L_ + m0) * L_;
  const _Float16* Vb = Vs + (size_t)b * L_ * DX_ + h * DKV_;

  v8f acc0 = {}, acc1 = {}, acc2 = {}, acc3 = {};
  for (int k0 = 0; k0 < L_; k0 += 32) {
    __builtin_prefetch(Ab + k0 + 32, 0, 1);
    v16h a  = frag_a_f32(Ab + k0, L_);
    const _Float16* Bk = Vb + (size_t)k0 * DX_;
    v16h b0 = frag_b_f16(Bk +  0, DX_);
    v16h b1 = frag_b_f16(Bk + 16, DX_);
    v16h b2 = frag_b_f16(Bk + 32, DX_);
    v16h b3 = frag_b_f16(Bk + 48, DX_);
    acc0 = wmma_f16(a, b0, acc0);
    acc1 = wmma_f16(a, b1, acc1);
    acc2 = wmma_f16(a, b2, acc2);
    acc3 = wmma_f16(a, b3, acc3);
  }
  const int lane = threadIdx.x & 31, half = lane >> 4, col = lane & 15;
  const v8f* accs[4] = {&acc0, &acc1, &acc2, &acc3};
#pragma unroll
  for (int t = 0; t < 4; ++t) {
#pragma unroll
    for (int r = 0; r < 8; ++r) {
      const int m = (half << 3) + r;
      ctx[((size_t)b * L_ + m0 + m) * DX_ + h * DKV_ + t * 16 + col] =
          (_Float16)((*accs[t])[r]);
    }
  }
}

// xo = ctx @ Wo + bo (f32 out), B fragments from pre-transposed f16 WoT.
__global__ void oproj_kernel(const _Float16* __restrict__ ctx,
                             const _Float16* __restrict__ WoT,
                             const float* __restrict__ bo,
                             float* __restrict__ xo) {
  const int wave = threadIdx.x >> 5;
  const int lin  = blockIdx.x * 8 + wave;      // 0..8191
  const int m0   = (lin >> 4) << 4;
  const int n0   = (lin & 15) << 6;

  v8f acc0 = {}, acc1 = {}, acc2 = {}, acc3 = {};
  for (int k0 = 0; k0 < DX_; k0 += 32) {
    const _Float16* Arow = ctx + (size_t)m0 * DX_ + k0;
    __builtin_prefetch(Arow + 32, 0, 1);
    v16h a  = frag_a_f16(Arow, DX_);
    // B[k][n] = WoT[(n0+n)*1024 + k0+k] : contiguous 32B per lane
    v16h b0 = frag_bT_f16(WoT + (size_t)(n0 +  0) * DX_ + k0, DX_);
    v16h b1 = frag_bT_f16(WoT + (size_t)(n0 + 16) * DX_ + k0, DX_);
    v16h b2 = frag_bT_f16(WoT + (size_t)(n0 + 32) * DX_ + k0, DX_);
    v16h b3 = frag_bT_f16(WoT + (size_t)(n0 + 48) * DX_ + k0, DX_);
    acc0 = wmma_f16(a, b0, acc0);
    acc1 = wmma_f16(a, b1, acc1);
    acc2 = wmma_f16(a, b2, acc2);
    acc3 = wmma_f16(a, b3, acc3);
  }
  const int lane = threadIdx.x & 31, half = lane >> 4, col = lane & 15;
  const v8f* accs[4] = {&acc0, &acc1, &acc2, &acc3};
#pragma unroll
  for (int t = 0; t < 4; ++t) {
    const float bb = bo[n0 + t * 16 + col];
#pragma unroll
    for (int r = 0; r < 8; ++r) {
      const int m = (half << 3) + r;
      xo[(size_t)(m0 + m) * DX_ + n0 + t * 16 + col] = (*accs[t])[r] + bb;
    }
  }
}

// y = LayerNorm(xo + Q) * gamma + beta ; one block per 1024-wide row
__global__ void residual_layernorm(const float* __restrict__ xo,
                                   const float* __restrict__ Q,
                                   const float* __restrict__ gamma,
                                   const float* __restrict__ beta,
                                   float* __restrict__ y) {
  __shared__ float rs[256], rs2[256];
  const int tid = threadIdx.x;
  const size_t row = blockIdx.x;
  const float* xr = xo + row * DX_;
  const float* qr = Q  + row * DX_;

  float xv[4];
  float s = 0.f, s2 = 0.f;
#pragma unroll
  for (int j = 0; j < 4; ++j) {
    const int c = tid + j * 256;
    const float x = xr[c] + qr[c];
    xv[j] = x; s += x; s2 += x * x;
  }
  rs[tid] = s; rs2[tid] = s2; __syncthreads();
  for (int st = 128; st > 0; st >>= 1) {
    if (tid < st) { rs[tid] += rs[tid + st]; rs2[tid] += rs2[tid + st]; }
    __syncthreads();
  }
  const float mu  = rs[0] * (1.f / DX_);
  const float var = rs2[0] * (1.f / DX_) - mu * mu;
  const float inv = rsqrtf(var + EPS_);

  float* yr = y + row * DX_;
#pragma unroll
  for (int j = 0; j < 4; ++j) {
    const int c = tid + j * 256;
    yr[c] = (xv[j] - mu) * inv * gamma[c] + beta[c];
  }
}

// ---------------- launcher ----------------
extern "C" void kernel_launch(void* const* d_in, const int* in_sizes, int n_in,
                              void* d_out, int out_size, void* d_ws, size_t ws_size,
                              hipStream_t stream) {
  (void)in_sizes; (void)n_in; (void)out_size; (void)ws_size;

  const float* Q  = (const float*)d_in[0];
  const float* K  = (const float*)d_in[1];
  const float* V  = (const float*)d_in[2];
  const unsigned char* mask = (const unsigned char*)d_in[3];
  const float* Wq = (const float*)d_in[4];
  const float* bq = (const float*)d_in[5];
  const float* Wk = (const float*)d_in[6];
  const float* bk = (const float*)d_in[7];
  const float* Wv = (const float*)d_in[8];
  const float* bv = (const float*)d_in[9];
  const float* Wo = (const float*)d_in[10];
  const float* bo = (const float*)d_in[11];
  const float* gamma = (const float*)d_in[12];
  const float* beta  = (const float*)d_in[13];

  float* y   = (float*)d_out;                          // [B,L,DX]
  float* att = y + (size_t)B_ * L_ * DX_;              // [H*B, L, L]

  // workspace carve-up (~104 MB total)
  char* ws = (char*)d_ws;
  const size_t ne = (size_t)B_ * L_ * (H_ * DKV_);     // 8 Mi elements
  const size_t nw = (size_t)DX_ * DX_;                 // 1 Mi elements
  _Float16* Qs  = (_Float16*)ws; ws += ne * sizeof(_Float16);
  _Float16* Ks  = (_Float16*)ws; ws += ne * sizeof(_Float16);
  _Float16* Vs  = (_Float16*)ws; ws += ne * sizeof(_Float16);
  _Float16* ctx = (_Float16*)ws; ws += ne * sizeof(_Float16);
  float*    xo  = (float*)ws;    ws += ne * sizeof(float);
  _Float16* WqT = (_Float16*)ws; ws += nw * sizeof(_Float16);
  _Float16* WkT = (_Float16*)ws; ws += nw * sizeof(_Float16);
  _Float16* WvT = (_Float16*)ws; ws += nw * sizeof(_Float16);
  _Float16* WoT = (_Float16*)ws; ws += nw * sizeof(_Float16);

  const dim3 blk(256);
  transpose_w_f16<<<dim3(32, 32), blk, 0, stream>>>(Wq, WqT);
  transpose_w_f16<<<dim3(32, 32), blk, 0, stream>>>(Wk, WkT);
  transpose_w_f16<<<dim3(32, 32), blk, 0, stream>>>(Wv, WvT);
  transpose_w_f16<<<dim3(32, 32), blk, 0, stream>>>(Wo, WoT);

  qkv_proj<<<dim3(1024, 3), blk, 0, stream>>>(Q, K, V, WqT, WkT, WvT,
                                              bq, bk, bv, Qs, Ks, Vs);
  fused_scores_softmax<<<dim3(L_ / 16, B_ * H_), blk, 0, stream>>>(Qs, Ks, mask,
                                                                   att);
  attv_kernel<<<dim3(16, B_ * H_), blk, 0, stream>>>(att, Vs, ctx);
  oproj_kernel<<<dim3(1024), blk, 0, stream>>>(ctx, WoT, bo, xo);
  residual_layernorm<<<dim3(B_ * L_), blk, 0, stream>>>(xo, Q, gamma, beta, y);
}
